// UpCat_Att_72258529788739
// MI455X (gfx1250) — matrix operand order repaired and checked by reading, not compiled, not run on hardware
//
#include <hip/hip_runtime.h>
#include <hip/hip_bf16.h>

typedef __attribute__((ext_vector_type(16))) _Float16 v16h;
typedef __attribute__((ext_vector_type(8)))  _Float16 v8h;
typedef __attribute__((ext_vector_type(8)))  float    v8f;

#define DEV static __device__ __forceinline__

static constexpr long long NVOX = 110592;      // 48*48*48
static constexpr long long NB   = 4LL * NVOX;  // batch*voxels = 442368

// ---------------------------------------------------------------------------
// WMMA fragment helpers (V_WMMA_F32_16X16X32_F16, wave32)
// A (16x32 f16): lane m(0-15): elems 0-15 = K k0+0..15
//                lane m+16   : elems 0-15 = K k0+16..31
// B (32x16 f16): lane n(0-15) holds K k0+0..15 of column n; lane n+16 holds K k0+16..31
// C/D (16x16 f32): lane L: col = L&15, rows (L>>4)*8 .. +7 in acc[0..7]
// ---------------------------------------------------------------------------

DEV v8f wmma_f16(v16h a, v16h b, v8f c) {
  return __builtin_amdgcn_wmma_f32_16x16x32_f16(false, a, false, b, (short)0, c, false, false);
}

// Weights pre-converted to f16, row-major [O, ldw], ldw multiple of 32, zero-padded.
DEV v16h load_a_h(const _Float16* __restrict__ W, int ldw, int row0, int k0, int lane) {
  const _Float16* p = W + (long long)(row0 + (lane & 15)) * ldw + k0 + ((lane >> 4) << 4);
  v8h lo = *(const v8h*)p;         // K kb+0..7
  v8h hi = *(const v8h*)(p + 8);   // K kb+8..15
  v16h a;
#pragma unroll
  for (int j = 0; j < 8; ++j) { a[j] = lo[j]; a[8 + j] = hi[j]; }
  return a;
}

// Activations: channels-last f16 [row, ldx], column = voxel; zero beyond Kreal
DEV v16h load_b_x(const _Float16* __restrict__ X, int ldx, int Kreal,
                  long long n0, int k0, int lane) {
  int ksel = (lane >> 4) << 4;
  v16h b;
  if (k0 + ksel < Kreal) {
    b = *(const v16h*)(X + (n0 + (lane & 15)) * (long long)ldx + k0 + ksel);
  } else {
#pragma unroll
    for (int j = 0; j < 16; ++j) b[j] = (_Float16)0.f;
  }
  return b;
}

// ---------------------------------------------------------------------------
// Weight pre-conversion (runs once per launch; deterministic)
// f32 [O,K] -> f16 [O,Kp], zero-padded
// ---------------------------------------------------------------------------
__global__ void convert_w(const float* __restrict__ W, _Float16* __restrict__ Wh,
                          int O, int K, int Kp) {
  int i = blockIdx.x * blockDim.x + threadIdx.x;
  int total = O * Kp;
  if (i >= total) return;
  int o = i / Kp, k = i % Kp;
  Wh[i] = (k < K) ? (_Float16)W[o * K + k] : (_Float16)0.f;
}

// conv3d weights f32 [32][CIN][27] -> f16 tap-major [27][32][CIN]
__global__ void convert_cw(const float* __restrict__ W, _Float16* __restrict__ Wh, int CIN) {
  int i = blockIdx.x * blockDim.x + threadIdx.x;
  int total = 32 * CIN * 27;
  if (i >= total) return;
  int tap = i % 27;
  int c = (i / 27) % CIN;
  int o = i / (27 * CIN);
  Wh[(tap * 32 + o) * CIN + c] = (_Float16)W[i];
}

// ---------------------------------------------------------------------------
// Generic 1x1x1 conv as GEMM: Y[n, o] = X[n, :K] @ W[o, :K]^T + bias[o]
// One wave per 16-voxel tile; Ntot = NB (divisible by 16).
// Wh is f16, padded to KT = roundup32(K). WF/WH select f32/f16 outputs at
// compile time (straight-line, vectorizable epilogue).
// ---------------------------------------------------------------------------
template <int K, int O, bool WF, bool WH>
__global__ void gemm_wmma(const _Float16* __restrict__ X, int ldx,
                          const _Float16* __restrict__ Wh, const float* __restrict__ bias,
                          float* __restrict__ Yf, _Float16* __restrict__ Yh) {
  int lane = threadIdx.x & 31;
  long long tile = (long long)blockIdx.x * (blockDim.x >> 5) + (threadIdx.x >> 5);
  long long n0 = tile * 16;
  if (n0 >= NB) return;  // wave-uniform
  constexpr int MT = O / 16;
  constexpr int KT = ((K + 31) / 32) * 32;

  __builtin_prefetch(X + (n0 + (lane & 15)) * (long long)ldx, 0, 3);

  v8f acc[MT];
#pragma unroll
  for (int t = 0; t < MT; ++t)
#pragma unroll
    for (int j = 0; j < 8; ++j) acc[t][j] = 0.f;

#pragma unroll
  for (int k0 = 0; k0 < KT; k0 += 32) {
    v16h bfrag = load_b_x(X, ldx, K, n0, k0, lane);
#pragma unroll
    for (int t = 0; t < MT; ++t) {
      v16h afrag = load_a_h(Wh, KT, t * 16, k0, lane);
      acc[t] = wmma_f16(afrag, bfrag, acc[t]);
    }
  }

  long long row = n0 + (lane & 15);
  int mb = (lane >> 4) * 8;
#pragma unroll
  for (int t = 0; t < MT; ++t)
#pragma unroll
    for (int r = 0; r < 8; ++r) {
      int m = t * 16 + mb + r;
      float v = acc[t][r] + bias[m];
      if constexpr (WF) Yf[row * O + m] = v;
      if constexpr (WH) Yh[row * O + m] = (_Float16)v;
    }
}

// ---------------------------------------------------------------------------
// 3x3x3 conv, pad=1, O=32, implicit GEMM via WMMA.
// Wh: f16 tap-major [27][32][CIN]. X: f16 channels-last [B, 48^3, CIN].
// One wave = 16 voxels along W-axis x 32 output channels.
// ---------------------------------------------------------------------------
template <int CIN>
__global__ void conv3d_wmma(const _Float16* __restrict__ X, const _Float16* __restrict__ Wh,
                            const float* __restrict__ bias, float* __restrict__ Yf) {
  const long long NTILES = 4LL * 48 * 48 * 3;
  int lane = threadIdx.x & 31;
  long long tile = (long long)blockIdx.x * (blockDim.x >> 5) + (threadIdx.x >> 5);
  if (tile >= NTILES) return;  // wave-uniform
  int wt = (int)(tile % 3);
  long long r = tile / 3;
  int h = (int)(r % 48); r /= 48;
  int d = (int)(r % 48);
  int b = (int)(r / 48);

  int col  = lane & 15;
  int ksel = (lane >> 4) << 4;

  __builtin_prefetch(X + (((long long)b * NVOX) + ((long long)d * 48 + h) * 48 + wt * 16 + col) * CIN, 0, 3);

  v8f acc0, acc1;
#pragma unroll
  for (int j = 0; j < 8; ++j) { acc0[j] = 0.f; acc1[j] = 0.f; }

  for (int dd = -1; dd <= 1; ++dd)
    for (int dh = -1; dh <= 1; ++dh)
#pragma unroll
      for (int dw = -1; dw <= 1; ++dw) {
        int z = d + dd, y = h + dh, x = wt * 16 + col + dw;
        bool ok = ((unsigned)z < 48u) && ((unsigned)y < 48u) && ((unsigned)x < 48u);
        const _Float16* px =
            X + (((long long)b * NVOX) + ((long long)z * 48 + y) * 48 + x) * CIN + ksel;
        int tap = (dd + 1) * 9 + (dh + 1) * 3 + (dw + 1);
        const _Float16* wt16 = Wh + (long long)tap * 32 * CIN;
#pragma unroll
        for (int k0 = 0; k0 < CIN; k0 += 32) {
          v16h bf;
          if (ok) {
            bf = *(const v16h*)(px + k0);
          } else {
#pragma unroll
            for (int j = 0; j < 16; ++j) bf[j] = (_Float16)0.f;
          }
          v16h a0 = load_a_h(wt16, CIN, 0, k0, lane);
          v16h a1 = load_a_h(wt16, CIN, 16, k0, lane);
          acc0 = wmma_f16(a0, bf, acc0);
          acc1 = wmma_f16(a1, bf, acc1);
        }
      }

  long long n = ((long long)b * NVOX) + ((long long)d * 48 + h) * 48 + wt * 16 + col;
  int mb = (lane >> 4) * 8;
#pragma unroll
  for (int r8 = 0; r8 < 8; ++r8) {
    Yf[n * 32 + mb + r8]      = acc0[r8] + bias[mb + r8];
    Yf[n * 32 + 16 + mb + r8] = acc1[r8] + bias[16 + mb + r8];
  }
}

// ---------------------------------------------------------------------------
// Stage 0: build concat input, channels-last f16: xcat[b,n,0:32]=x_e, [32:64]=deconv(x)
// ---------------------------------------------------------------------------
__global__ void prep_xe_kernel(const float* __restrict__ XE, _Float16* __restrict__ xcat) {
  long long i = (long long)blockIdx.x * blockDim.x + threadIdx.x;
  const long long total = 4LL * 32 * NVOX;
  if (i >= total) return;
  int n = (int)(i % NVOX);
  int c = (int)((i / NVOX) % 32);
  int b = (int)(i / (NVOX * 32));
  xcat[((long long)b * NVOX + n) * 64 + c] = (_Float16)XE[i];
}

__global__ void deconv_kernel(const float* __restrict__ X, const float* __restrict__ Wt,
                              const float* __restrict__ Bt, _Float16* __restrict__ xcat) {
  long long i = (long long)blockIdx.x * blockDim.x + threadIdx.x;
  const long long total = 4LL * NVOX * 32;
  if (i >= total) return;
  int o = (int)(i % 32);
  long long r = i / 32;
  int n = (int)(r % NVOX);
  int b = (int)(r / NVOX);
  int w = n % 48, h = (n / 48) % 48, d = n / 2304;
  int wi = w >> 1, hi = h >> 1, di = d >> 1;
  int pidx = ((d & 1) * 2 + (h & 1)) * 2 + (w & 1);
  long long vin = ((long long)di * 24 + hi) * 24 + wi;
  const float* xb = X + (long long)b * 64 * 13824;
  float s = Bt[o];
#pragma unroll 8
  for (int c = 0; c < 64; ++c)
    s += xb[(long long)c * 13824 + vin] * Wt[((long long)c * 32 + o) * 8 + pidx];
  xcat[((long long)b * NVOX + n) * 64 + 32 + o] = (_Float16)s;
}

// ---------------------------------------------------------------------------
// Moments (sum, sumsq) per (batch, group) with float atomics.
// X: f32 [B*NVOX, C], groups of GS consecutive channels.
// ---------------------------------------------------------------------------
__global__ void zero_f32(float* p, int n) {
  int i = blockIdx.x * blockDim.x + threadIdx.x;
  if (i < n) p[i] = 0.f;
}

__global__ void moments_reduce(const float* __restrict__ X, float* __restrict__ st,
                               int C, int GS, int chunks) {
  int G = C / GS;
  int bg = blockIdx.x / chunks;
  int chunk = blockIdx.x % chunks;
  int b = bg / G, g = bg % G;
  long long per = (NVOX + chunks - 1) / chunks;
  long long n0 = (long long)chunk * per;
  long long n1 = n0 + per; if (n1 > NVOX) n1 = NVOX;
  long long cnt = (n1 - n0) * GS;
  float s = 0.f, s2 = 0.f;
  for (long long idx = threadIdx.x; idx < cnt; idx += blockDim.x) {
    long long n = n0 + idx / GS;
    int c = g * GS + (int)(idx % GS);
    float v = X[((long long)b * NVOX + n) * C + c];
    s += v; s2 += v * v;
  }
  __shared__ float rs[256], rq[256];
  int t = threadIdx.x;
  rs[t] = s; rq[t] = s2;
  __syncthreads();
  for (int off = 128; off > 0; off >>= 1) {
    if (t < off) { rs[t] += rs[t + off]; rq[t] += rq[t + off]; }
    __syncthreads();
  }
  if (t == 0) {
    atomicAdd(&st[bg * 2 + 0], rs[0]);
    atomicAdd(&st[bg * 2 + 1], rq[0]);
  }
}

// normalize + affine(optional) + leaky, write f16 channels-last
__global__ void gn_apply(const float* __restrict__ X, const float* __restrict__ st,
                         const float* __restrict__ gamma, const float* __restrict__ beta,
                         _Float16* __restrict__ Yh, int C, int GS, float slope, float invcnt) {
  long long i = (long long)blockIdx.x * blockDim.x + threadIdx.x;
  long long total = NB * C;
  if (i >= total) return;
  int c = (int)(i % C);
  long long row = i / C;
  int b = (int)(row / NVOX);
  int G = C / GS;
  int g = c / GS;
  float mean = st[(b * G + g) * 2 + 0] * invcnt;
  float var  = st[(b * G + g) * 2 + 1] * invcnt - mean * mean;
  float v = (X[i] - mean) * rsqrtf(var + 1e-5f);
  if (gamma) v = v * gamma[c] + beta[c];
  v = (v >= 0.f) ? v : slope * v;
  Yh[i] = (_Float16)v;
}

// final instance-norm + leaky(0.01), channels-last f32 -> NCDHW f32 output
__global__ void in_apply_out(const float* __restrict__ X, const float* __restrict__ st,
                             float* __restrict__ out) {
  long long i = (long long)blockIdx.x * blockDim.x + threadIdx.x;
  const long long total = 4LL * 32 * NVOX;
  if (i >= total) return;
  int n = (int)(i % NVOX);
  int c = (int)((i / NVOX) % 32);
  int b = (int)(i / (NVOX * 32));
  const float invc = 1.f / 110592.f;
  float mean = st[(b * 32 + c) * 2 + 0] * invc;
  float var  = st[(b * 32 + c) * 2 + 1] * invc - mean * mean;
  float v = (X[((long long)b * NVOX + n) * 32 + c] - mean) * rsqrtf(var + 1e-5f);
  out[i] = (v >= 0.f) ? v : 0.01f * v;
}

// ---------------------------------------------------------------------------
// pool_sum: block max+avg reduce 48^3 -> 6^3 (factor 8), channels-last f16 in, f32 out
// ---------------------------------------------------------------------------
__global__ void pool_kernel(const _Float16* __restrict__ X, float* __restrict__ P) {
  int cell = blockIdx.x % 216;
  int b = blockIdx.x / 216;
  int c = threadIdx.x;
  if (c >= 48) return;
  int ow = cell % 6, oh = (cell / 6) % 6, od = cell / 36;
  float mx = -3.0e38f, sm = 0.f;
  for (int fd = 0; fd < 8; ++fd)
    for (int fh = 0; fh < 8; ++fh)
      for (int fw = 0; fw < 8; ++fw) {
        int n = ((od * 8 + fd) * 48 + oh * 8 + fh) * 48 + ow * 8 + fw;
        float v = (float)X[((long long)b * NVOX + n) * 48 + c];
        mx = fmaxf(mx, v);
        sm += v;
      }
  P[((long long)b * 216 + cell) * 48 + c] = mx + sm * (1.f / 512.f);
}

// ---------------------------------------------------------------------------
// Linear cross-attention on 216 tokens, C=48, 8 heads of 6. One block per batch.
// Q, KV: f32 [B,216,48]. A out: f32 [B,48,216].
// ---------------------------------------------------------------------------
__global__ void attn_kernel(const float* __restrict__ Q, const float* __restrict__ KV,
                            const float* __restrict__ wq, const float* __restrict__ bq,
                            const float* __restrict__ wk, const float* __restrict__ bk,
                            const float* __restrict__ wv, const float* __restrict__ bv,
                            const float* __restrict__ wo, const float* __restrict__ bo,
                            float* __restrict__ A) {
  __shared__ _Float16 kh_s[216 * 48];
  __shared__ _Float16 vh_s[216 * 48];
  __shared__ float mx_s[48], is_s[48];
  __shared__ float ctx_s[288];
  int b = blockIdx.x, tid = threadIdx.x;
  const float* kvb = KV + (long long)b * 216 * 48;

  if (tid < 216) {
    float tok[48];
    for (int c = 0; c < 48; ++c) tok[c] = kvb[tid * 48 + c];
    for (int o = 0; o < 48; ++o) {
      float sk = bk[o], sv = bv[o];
      for (int c = 0; c < 48; ++c) {
        sk += tok[c] * wk[o * 48 + c];
        sv += tok[c] * wv[o * 48 + c];
      }
      kh_s[tid * 48 + o] = (_Float16)sk;
      vh_s[tid * 48 + o] = (_Float16)sv;
    }
  }
  __syncthreads();

  if (tid < 48) {  // softmax over tokens for each (h,d) column
    float mx = -3.0e38f;
    for (int n = 0; n < 216; ++n) mx = fmaxf(mx, (float)kh_s[n * 48 + tid]);
    float s = 0.f;
    for (int n = 0; n < 216; ++n) s += __expf((float)kh_s[n * 48 + tid] - mx);
    mx_s[tid] = mx;
    is_s[tid] = 1.f / s;
  }
  __syncthreads();

  if (tid < 288) {  // ctx[h][d][e] = sum_n ks * vh
    int hh = tid / 36, dd = (tid % 36) / 6, ee = tid % 6;
    int cd = hh * 6 + dd, ce = hh * 6 + ee;
    float mx = mx_s[cd], inv = is_s[cd];
    float s = 0.f;
    for (int n = 0; n < 216; ++n)
      s += __expf((float)kh_s[n * 48 + cd] - mx) * inv * (float)vh_s[n * 48 + ce];
    ctx_s[tid] = s;
  }
  __syncthreads();

  if (tid < 216) {
    const float* qt = Q + ((long long)b * 216 + tid) * 48;
    float tok[48];
    for (int c = 0; c < 48; ++c) tok[c] = qt[c];
    float outv[48];
    for (int hh = 0; hh < 8; ++hh) {
      float qh[6];
      float mx = -3.0e38f;
      for (int d2 = 0; d2 < 6; ++d2) {
        int o = hh * 6 + d2;
        float s = bq[o];
        for (int c = 0; c < 48; ++c) s += tok[c] * wq[o * 48 + c];
        qh[d2] = s;
        mx = fmaxf(mx, s);
      }
      float se = 0.f;
      for (int d2 = 0; d2 < 6; ++d2) { qh[d2] = __expf(qh[d2] - mx); se += qh[d2]; }
      float inv = 1.f / se;
      for (int ee = 0; ee < 6; ++ee) {
        float s = 0.f;
        for (int d2 = 0; d2 < 6; ++d2) s += qh[d2] * ctx_s[(hh * 6 + d2) * 6 + ee];
        outv[hh * 6 + ee] = s * inv;
      }
    }
    for (int o = 0; o < 48; ++o) {
      float s = bo[o];
      for (int c = 0; c < 48; ++c) s += outv[c] * wo[o * 48 + c];
      A[((long long)b * 48 + o) * 216 + tid] = s;
    }
  }
}

// ---------------------------------------------------------------------------
// Trilinear upsample 6^3 -> 48^3 (align_corners), sigmoid gate, multiply, concat
// A1/A2: [B,48,216] f32; I1/I2: [B,NVOX,48] f16; OC: [B,NVOX,96] f16
// ---------------------------------------------------------------------------
__global__ void gate_kernel(const float* __restrict__ A1, const float* __restrict__ A2,
                            const _Float16* __restrict__ I1, const _Float16* __restrict__ I2,
                            _Float16* __restrict__ OC) {
  long long i = (long long)blockIdx.x * blockDim.x + threadIdx.x;
  const long long total = NB * 96;
  if (i >= total) return;
  int j = (int)(i % 96);
  long long r = i / 96;
  int n = (int)(r % NVOX);
  int b = (int)(r / NVOX);
  int w = n % 48, h = (n / 48) % 48, d = n / 2304;
  const float* A = (j < 48) ? A1 : A2;
  const _Float16* I = (j < 48) ? I1 : I2;
  int c = (j < 48) ? j : j - 48;

  const float sc = 5.0f / 47.0f;
  float pd = d * sc, ph = h * sc, pw = w * sc;
  int d0 = (int)pd, h0 = (int)ph, w0 = (int)pw;
  int d1 = min(d0 + 1, 5), h1 = min(h0 + 1, 5), w1 = min(w0 + 1, 5);
  float fd = pd - d0, fh = ph - h0, fw = pw - w0;
  const float* ap = A + ((long long)b * 48 + c) * 216;
  auto V = [&](int z, int y, int x) { return ap[(z * 6 + y) * 6 + x]; };
  float c00 = V(d0, h0, w0) * (1.f - fw) + V(d0, h0, w1) * fw;
  float c01 = V(d0, h1, w0) * (1.f - fw) + V(d0, h1, w1) * fw;
  float c10 = V(d1, h0, w0) * (1.f - fw) + V(d1, h0, w1) * fw;
  float c11 = V(d1, h1, w0) * (1.f - fw) + V(d1, h1, w1) * fw;
  float v0 = c00 * (1.f - fh) + c01 * fh;
  float v1 = c10 * (1.f - fh) + c11 * fh;
  float val = v0 * (1.f - fd) + v1 * fd;
  float sig = 1.f / (1.f + __expf(-val));
  float xin = (float)I[((long long)b * NVOX + n) * 48 + c];
  OC[i] = (_Float16)(sig * xin);
}

// ---------------------------------------------------------------------------
extern "C" void kernel_launch(void* const* d_in, const int* in_sizes, int n_in,
                              void* d_out, int out_size, void* d_ws, size_t ws_size,
                              hipStream_t stream) {
  (void)in_sizes; (void)n_in; (void)out_size; (void)ws_size;

  const float* x     = (const float*)d_in[0];
  const float* x_e   = (const float*)d_in[1];
  const float* up_w  = (const float*)d_in[2];
  const float* up_b  = (const float*)d_in[3];
  const float* wf    = (const float*)d_in[4];
  const float* bfv   = (const float*)d_in[5];
  const float* w1    = (const float*)d_in[6];
  const float* b1    = (const float*)d_in[7];
  const float* w2    = (const float*)d_in[8];
  const float* b2    = (const float*)d_in[9];
  const float* w3    = (const float*)d_in[10];
  const float* b3    = (const float*)d_in[11];
  const float* w4    = (const float*)d_in[12];
  const float* b4    = (const float*)d_in[13];
  const float* gn_g  = (const float*)d_in[14];
  const float* gn_b  = (const float*)d_in[15];
  const float* a1w[8] = {(const float*)d_in[16], (const float*)d_in[17],
                         (const float*)d_in[18], (const float*)d_in[19],
                         (const float*)d_in[20], (const float*)d_in[21],
                         (const float*)d_in[22], (const float*)d_in[23]};
  const float* a2w[8] = {(const float*)d_in[24], (const float*)d_in[25],
                         (const float*)d_in[26], (const float*)d_in[27],
                         (const float*)d_in[28], (const float*)d_in[29],
                         (const float*)d_in[30], (const float*)d_in[31]};
  const float* wfin  = (const float*)d_in[32];
  const float* bfin  = (const float*)d_in[33];
  const float* c0_w  = (const float*)d_in[34];
  const float* c0_b  = (const float*)d_in[35];
  const float* c1_w  = (const float*)d_in[36];
  const float* c1_b  = (const float*)d_in[37];

  // ---- workspace carve ----
  char* p = (char*)d_ws;
  auto carve = [&](size_t bytes) -> void* {
    void* q = (void*)p;
    p += (bytes + 255) & ~(size_t)255;
    return q;
  };
  _Float16* xcat = (_Float16*)carve(NB * 64 * 2);   // [B,N,64] (x_e | deconv)
  _Float16* fush = (_Float16*)carve(NB * 48 * 2);   // fusion conv out (for pooling)
  float*    T    = (float*)   carve(NB * 48 * 4);   // f32 GEMM scratch (reused)
  _Float16* Hh   = (_Float16*)carve(NB * 48 * 2);   // f16 scratch (reused)
  _Float16* i1h  = (_Float16*)carve(NB * 48 * 2);
  _Float16* i2h  = (_Float16*)carve(NB * 48 * 2);
  float* P1  = (float*)carve(4LL * 216 * 48 * 4);
  float* P2  = (float*)carve(4LL * 216 * 48 * 4);
  float* PKV = (float*)carve(4LL * 216 * 48 * 4);
  float* A1  = (float*)carve(4LL * 48 * 216 * 4);
  float* A2  = (float*)carve(4LL * 48 * 216 * 4);
  _Float16* OC = (_Float16*)carve(NB * 96 * 2);
  _Float16* FH = (_Float16*)carve(NB * 64 * 2);
  float* stG = (float*)carve(4LL * 6 * 2 * 4);
  float* stI = (float*)carve(4LL * 32 * 2 * 4);
  // f16 weight staging (WMMA-ready, padded)
  _Float16* wfh   = (_Float16*)carve(48 * 64 * 2);
  _Float16* w1h   = (_Float16*)carve(48 * 32 * 2);
  _Float16* w2h   = (_Float16*)carve(48 * 32 * 2);
  _Float16* w3h   = (_Float16*)carve(48 * 64 * 2);   // K=48 padded to 64
  _Float16* w4h   = (_Float16*)carve(48 * 64 * 2);   // K=48 padded to 64
  _Float16* wfinh = (_Float16*)carve(64 * 96 * 2);
  _Float16* c0h   = (_Float16*)carve(27LL * 32 * 64 * 2);
  _Float16* c1h   = (_Float16*)carve(27LL * 32 * 32 * 2);

  const int BLK = 256;
  const long long ELT32 = 4LL * 32 * NVOX;                 // 14,155,776
  const int gELT32 = (int)((ELT32 + BLK - 1) / BLK);
  const int gGEMM  = (int)(NB / 16 / 8);                   // 3456 blocks, 8 waves each
  const float GN_INV = 1.f / (8.f * 110592.f);
  const float IN_INV = 1.f / 110592.f;

  // Stage -1: weight conversion to f16 (tiny)
  convert_w<<<(48 * 64 + BLK - 1) / BLK, BLK, 0, stream>>>(wf, wfh, 48, 64, 64);
  convert_w<<<(48 * 32 + BLK - 1) / BLK, BLK, 0, stream>>>(w1, w1h, 48, 32, 32);
  convert_w<<<(48 * 32 + BLK - 1) / BLK, BLK, 0, stream>>>(w2, w2h, 48, 32, 32);
  convert_w<<<(48 * 64 + BLK - 1) / BLK, BLK, 0, stream>>>(w3, w3h, 48, 48, 64);
  convert_w<<<(48 * 64 + BLK - 1) / BLK, BLK, 0, stream>>>(w4, w4h, 48, 48, 64);
  convert_w<<<(64 * 96 + BLK - 1) / BLK, BLK, 0, stream>>>(wfin, wfinh, 64, 96, 96);
  convert_cw<<<(27 * 32 * 64 + BLK - 1) / BLK, BLK, 0, stream>>>(c0_w, c0h, 64);
  convert_cw<<<(27 * 32 * 32 + BLK - 1) / BLK, BLK, 0, stream>>>(c1_w, c1h, 32);

  // Stage 0: staging
  prep_xe_kernel<<<gELT32, BLK, 0, stream>>>(x_e, xcat);
  deconv_kernel<<<gELT32, BLK, 0, stream>>>(x, up_w, up_b, xcat);

  // fusion = conv1x1(concat, 64->48)  (raw, for KV pooling)
  gemm_wmma<64, 48, false, true><<<gGEMM, BLK, 0, stream>>>(xcat, 64, wfh, bfv, nullptr, fush);

  // i1 chain: conv(x0,32->48) -> GN -> leaky -> conv(48->48) -> GN -> leaky
  gemm_wmma<32, 48, true, false><<<gGEMM, BLK, 0, stream>>>(xcat + 32, 64, w1h, b1, T, nullptr);
  zero_f32<<<1, 64, 0, stream>>>(stG, 48);
  moments_reduce<<<4 * 6 * 64, BLK, 0, stream>>>(T, stG, 48, 8, 64);
  gn_apply<<<(int)((NB * 48 + BLK - 1) / BLK), BLK, 0, stream>>>(T, stG, gn_g, gn_b, Hh, 48, 8, 0.1f, GN_INV);
  gemm_wmma<48, 48, true, false><<<gGEMM, BLK, 0, stream>>>(Hh, 48, w3h, b3, T, nullptr);
  zero_f32<<<1, 64, 0, stream>>>(stG, 48);
  moments_reduce<<<4 * 6 * 64, BLK, 0, stream>>>(T, stG, 48, 8, 64);
  gn_apply<<<(int)((NB * 48 + BLK - 1) / BLK), BLK, 0, stream>>>(T, stG, gn_g, gn_b, i1h, 48, 8, 0.1f, GN_INV);

  // i2 chain: conv(x_e,32->48) -> GN -> leaky -> conv(48->48) -> GN -> leaky
  gemm_wmma<32, 48, true, false><<<gGEMM, BLK, 0, stream>>>(xcat, 64, w2h, b2, T, nullptr);
  zero_f32<<<1, 64, 0, stream>>>(stG, 48);
  moments_reduce<<<4 * 6 * 64, BLK, 0, stream>>>(T, stG, 48, 8, 64);
  gn_apply<<<(int)((NB * 48 + BLK - 1) / BLK), BLK, 0, stream>>>(T, stG, gn_g, gn_b, Hh, 48, 8, 0.1f, GN_INV);
  gemm_wmma<48, 48, true, false><<<gGEMM, BLK, 0, stream>>>(Hh, 48, w4h, b4, T, nullptr);
  zero_f32<<<1, 64, 0, stream>>>(stG, 48);
  moments_reduce<<<4 * 6 * 64, BLK, 0, stream>>>(T, stG, 48, 8, 64);
  gn_apply<<<(int)((NB * 48 + BLK - 1) / BLK), BLK, 0, stream>>>(T, stG, gn_g, gn_b, i2h, 48, 8, 0.1f, GN_INV);

  // pooling to 6^3
  pool_kernel<<<4 * 216, 64, 0, stream>>>(i1h, P1);
  pool_kernel<<<4 * 216, 64, 0, stream>>>(i2h, P2);
  pool_kernel<<<4 * 216, 64, 0, stream>>>(fush, PKV);

  // linear attention (tiny)
  attn_kernel<<<4, BLK, 0, stream>>>(P1, PKV, a1w[0], a1w[1], a1w[2], a1w[3],
                                     a1w[4], a1w[5], a1w[6], a1w[7], A1);
  attn_kernel<<<4, BLK, 0, stream>>>(P2, PKV, a2w[0], a2w[1], a2w[2], a2w[3],
                                     a2w[4], a2w[5], a2w[6], a2w[7], A2);

  // upsample + sigmoid gate + concat
  gate_kernel<<<(int)((NB * 96 + BLK - 1) / BLK), BLK, 0, stream>>>(A1, A2, i1h, i2h, OC);

  // final 1x1 conv (96 -> 64)
  gemm_wmma<96, 64, false, true><<<gGEMM, BLK, 0, stream>>>(OC, 96, wfinh, bfin, nullptr, FH);

  // conv3x3x3 #1 (64 -> 32) + instance norm + leaky(0.01)
  conv3d_wmma<64><<<gGEMM, BLK, 0, stream>>>(FH, c0h, c0_b, T);
  zero_f32<<<1, BLK, 0, stream>>>(stI, 256);
  moments_reduce<<<4 * 32 * 64, BLK, 0, stream>>>(T, stI, 32, 1, 64);
  gn_apply<<<(int)((NB * 32 + BLK - 1) / BLK), BLK, 0, stream>>>(T, stI, nullptr, nullptr, Hh, 32, 1, 0.01f, IN_INV);

  // conv3x3x3 #2 (32 -> 32) + instance norm + leaky(0.01) -> NCDHW f32 out
  conv3d_wmma<32><<<gGEMM, BLK, 0, stream>>>(Hh, c1h, c1_b, T);
  zero_f32<<<1, BLK, 0, stream>>>(stI, 256);
  moments_reduce<<<4 * 32 * 64, BLK, 0, stream>>>(T, stI, 32, 1, 64);
  in_apply_out<<<gELT32, BLK, 0, stream>>>(T, stI, (float*)d_out);
}
